// GNN_86672440033854
// MI455X (gfx1250) — matrix operand order repaired and checked by reading, not compiled, not run on hardware
//
#include <hip/hip_runtime.h>
#include <hip/hip_bf16.h>
#include <stdint.h>

// ---------------------------------------------------------------------------
// Problem constants (match reference)
// ---------------------------------------------------------------------------
#define NN      10000     // nodes
#define EE      30000     // edges
#define NG      256       // graphs
#define HH      64        // hidden
#define INF_    11        // input features
#define EFF     4         // edge features
#define LLAY    4         // conv layers
#define MITER   3         // set2set iters

typedef __attribute__((ext_vector_type(16))) _Float16 v16h;
typedef __attribute__((ext_vector_type(8)))  float    v8f;

static __device__ __forceinline__ v8f wmma_f16(v16h a, v16h b, v8f c) {
  // D = A(16x32 f16) * B(32x16 f16) + C(16x16 f32)
  return __builtin_amdgcn_wmma_f32_16x16x32_f16(false, a, false, b, (short)0, c,
                                                false, false);
}

static __device__ __forceinline__ float dev_sigmoid(float x) {
  return 1.0f / (1.0f + __expf(-x));
}
static __device__ __forceinline__ float dev_silu(float x) {
  return x / (1.0f + __expf(-x));
}
// order-preserving float->uint encoding for atomicMax on signed floats
static __device__ __forceinline__ unsigned fenc(float f) {
  unsigned u = __float_as_uint(f);
  return (u & 0x80000000u) ? ~u : (u | 0x80000000u);
}
static __device__ __forceinline__ float fdec(unsigned u) {
  return __uint_as_float((u & 0x80000000u) ? (u & 0x7fffffffu) : ~u);
}

// ---------------------------------------------------------------------------
// Generic helpers
// ---------------------------------------------------------------------------
__global__ void zero_k(float* p, int n) {
  int t = blockIdx.x * 256 + threadIdx.x;
  if (t < n) p[t] = 0.0f;
}

// ---------------------------------------------------------------------------
// Weight repack: build WMMA B-operand fragment buffers (f16).
// B fragment layout assumption (mirror of documented A layout):
//   lane l: column n = l&15, element j holds K = (j&7) + 16*(j>>3) + 8*(l>>4).
// Buffer element index = frag*512 + l*16 + j  -> each lane loads 32B contig.
//   Bv    : frag = (i*4 + kc)*4 + nt        (i<64, kc<4, nt<4)   [We2 folded]
//   Wcatv : frag = kc*16 + nt               (kc<2, nt<16)  [Wroot^T | Whh^T]
//   Wihv  : frag = kc*12 + nt               (kc<2, nt<12)  [Wih^T]
// ---------------------------------------------------------------------------
#define BV_ELEMS    524288
#define WCAT_ELEMS  16384
#define WIH_ELEMS   12288
#define PREP_TOTAL  (BV_ELEMS + WCAT_ELEMS + WIH_ELEMS)

__global__ void prep_k(const float* __restrict__ We2,
                       const float* __restrict__ Wroot,
                       const float* __restrict__ Whh,
                       const float* __restrict__ Wih,
                       _Float16* __restrict__ Bv,
                       _Float16* __restrict__ Wcatv,
                       _Float16* __restrict__ Wihv) {
  int tid = blockIdx.x * 256 + threadIdx.x;
  if (tid < BV_ELEMS) {
    int j = tid & 15, l = (tid >> 4) & 31;
    int n = l & 15, K = (j & 7) + 16 * (j >> 3) + 8 * (l >> 4);
    int frag = tid >> 9;
    int nt = frag & 3, kc = (frag >> 2) & 3, i = frag >> 4;
    // Wv[i*128 + k, o] = We2[i*64 + o, k]
    int row = i * 64 + nt * 16 + n;
    int col = kc * 32 + K;
    Bv[tid] = (_Float16)We2[(size_t)row * 128 + col];
  } else if (tid < BV_ELEMS + WCAT_ELEMS) {
    int t2 = tid - BV_ELEMS;
    int j = t2 & 15, l = (t2 >> 4) & 31;
    int n = l & 15, K = (j & 7) + 16 * (j >> 3) + 8 * (l >> 4);
    int frag = t2 >> 9;
    int nt = frag & 15, kc = frag >> 4;
    int o = nt * 16 + n, k = kc * 32 + K;
    float v = (o < 64) ? Wroot[o * 64 + k] : Whh[(o - 64) * 64 + k];
    Wcatv[t2] = (_Float16)v;
  } else if (tid < PREP_TOTAL) {
    int t3 = tid - BV_ELEMS - WCAT_ELEMS;
    int j = t3 & 15, l = (t3 >> 4) & 31;
    int n = l & 15, K = (j & 7) + 16 * (j >> 3) + 8 * (l >> 4);
    int frag = t3 >> 9;
    int nt = frag % 12, kc = frag / 12;
    int o = nt * 16 + n, k = kc * 32 + K;
    Wihv[t3] = (_Float16)Wih[o * 64 + k];
  }
}

// ---------------------------------------------------------------------------
// h0 = silu(x @ W1^T + b1)
// ---------------------------------------------------------------------------
__global__ void embed_k(const float* __restrict__ x, const float* __restrict__ W1,
                        const float* __restrict__ b1, float* __restrict__ h) {
  int t = blockIdx.x * 256 + threadIdx.x;
  if (t >= NN * 64) return;
  int n = t >> 6, o = t & 63;
  float a = b1[o];
#pragma unroll
  for (int i = 0; i < INF_; ++i) a += x[n * INF_ + i] * W1[o * INF_ + i];
  h[t] = dev_silu(a);
}

// e_lin = edge_attr @ We1^T + be1     [E,128]
__global__ void edgelin_k(const float* __restrict__ ea, const float* __restrict__ We1,
                          const float* __restrict__ be1, float* __restrict__ elin) {
  int t = blockIdx.x * 256 + threadIdx.x;
  if (t >= EE * 128) return;
  int e = t >> 7, c = t & 127;
  float a = be1[c];
#pragma unroll
  for (int k = 0; k < EFF; ++k) a += ea[e * EFF + k] * We1[c * EFF + k];
  elin[t] = a;
}

// per-column batch stats over E rows (biased var, eps=1e-5)
__global__ void colstats_k(const float* __restrict__ elin,
                           float* __restrict__ mean, float* __restrict__ inv) {
  int c = blockIdx.x;              // 128 columns
  int tid = threadIdx.x;           // 256 threads
  float s = 0.0f, s2 = 0.0f;
  for (int e = tid; e < EE; e += 256) {
    float v = elin[(size_t)e * 128 + c];
    s += v; s2 += v * v;
  }
  __shared__ float bs[256], bs2[256];
  bs[tid] = s; bs2[tid] = s2;
  __syncthreads();
  for (int o = 128; o > 0; o >>= 1) {
    if (tid < o) { bs[tid] += bs[tid + o]; bs2[tid] += bs2[tid + o]; }
    __syncthreads();
  }
  if (tid == 0) {
    float mu = bs[0] / (float)EE;
    float var = bs2[0] / (float)EE - mu * mu;
    mean[c] = mu;
    inv[c] = rsqrtf(var + 1e-5f);
  }
}

// ehat = silu(bn(e_lin)) as f16   [E,128]
__global__ void bnact_k(const float* __restrict__ elin, const float* __restrict__ mean,
                        const float* __restrict__ inv, const float* __restrict__ g1,
                        const float* __restrict__ beta1, _Float16* __restrict__ eact) {
  int t = blockIdx.x * 256 + threadIdx.x;
  if (t >= EE * 128) return;
  int c = t & 127;
  float v = (elin[t] - mean[c]) * inv[c] * g1[c] + beta1[c];
  eact[t] = (_Float16)dev_silu(v);
}

// in-degree (float counts; clamped at use site)
__global__ void deg_k(const int* __restrict__ eidx, float* __restrict__ deg) {
  int e = blockIdx.x * 256 + threadIdx.x;
  if (e < EE) atomicAdd(&deg[eidx[EE + e]], 1.0f);
}

// nb[n,o] = sum_i h[n,i] * be2[i*64+o]   (theta bias contribution per source)
__global__ void nb_k(const float* __restrict__ h, const float* __restrict__ be2,
                     float* __restrict__ nb) {
  int t = blockIdx.x * 256 + threadIdx.x;
  if (t >= NN * 64) return;
  int n = t >> 6, o = t & 63;
  float a = 0.0f;
#pragma unroll 8
  for (int i = 0; i < 64; ++i) a += h[n * 64 + i] * be2[i * 64 + o];
  nb[t] = a;
}

// ---------------------------------------------------------------------------
// Fused NNConv edge kernel (the heavy WMMA kernel).
// One wave per 32-edge tile (two 16-row A tiles sharing every B fragment ->
// 1 global b128 per WMMA instead of 2; halves L2 B-stream traffic).
// msg[32x64] = A[32x8192] @ Wv[8192x64],
// A(e, i*128+k) = f16(xc[src_e,i]) * f16(ehat[e,k]) built on the fly.
// Accumulate into agg via global f32 atomics (mean divide happens in pass2).
// Tail tile: gather indices clamped, invalid rows flagged dst=-1 and skipped.
// ---------------------------------------------------------------------------
__global__ __launch_bounds__(32) void edge_msg_k(
    const float* __restrict__ h, const _Float16* __restrict__ eact,
    const _Float16* __restrict__ Bv, const float* __restrict__ nb,
    const int* __restrict__ eidx, float* __restrict__ agg) {
  __shared__ int srcs[32], dsts[32];
  __shared__ float xs[32 * 64];
  __shared__ float nbl[32 * 64];
  __shared__ _Float16 eh[32 * 128];

  const int lane = threadIdx.x;
  const int e0 = blockIdx.x * 32;
  {
    int e = e0 + lane;
    bool val = (e < EE);
    srcs[lane] = val ? eidx[e] : eidx[0];
    dsts[lane] = val ? eidx[EE + e] : -1;
  }
  __syncthreads();
  for (int r = 0; r < 32; ++r) {
    int s = srcs[r];
    xs[r * 64 + lane]       = h[(size_t)s * 64 + lane];
    xs[r * 64 + 32 + lane]  = h[(size_t)s * 64 + 32 + lane];
    nbl[r * 64 + lane]      = nb[(size_t)s * 64 + lane];
    nbl[r * 64 + 32 + lane] = nb[(size_t)s * 64 + 32 + lane];
  }
  {
    const uint32_t* ep = (const uint32_t*)eact;  // EE*64 dwords total
    uint32_t* eq = (uint32_t*)eh;
    const size_t lim = (size_t)EE * 64;
#pragma unroll
    for (int t = 0; t < 64; ++t) {
      size_t g = (size_t)e0 * 64 + t * 32 + lane;
      eq[t * 32 + lane] = (g < lim) ? ep[g] : 0u;
    }
  }
  __syncthreads();

  const int m = lane & 15;
  const int hi = lane >> 4;

  // Register-cache each lane's needed half of the two ehat rows it owns:
  // erG[kc*16 + j] = ehat[G*16 + m][kc*32 + K(j,hi)],
  // K = (j&7) + 16*(j>>3) + 8*hi.
  _Float16 er0[64], er1[64];
  {
    const uint32_t* eru0 = (const uint32_t*)eh + m * 64;
    const uint32_t* eru1 = (const uint32_t*)eh + (16 + m) * 64;
#pragma unroll
    for (int kc = 0; kc < 4; ++kc) {
#pragma unroll
      for (int t = 0; t < 8; ++t) {
        int Kb = 2 * t + ((t >= 4) ? 8 : 0) + 8 * hi;  // even
        union { uint32_t u; _Float16 f[2]; } cv;
        cv.u = eru0[(kc * 32 + Kb) >> 1];
        er0[kc * 16 + 2 * t]     = cv.f[0];
        er0[kc * 16 + 2 * t + 1] = cv.f[1];
        cv.u = eru1[(kc * 32 + Kb) >> 1];
        er1[kc * 16 + 2 * t]     = cv.f[0];
        er1[kc * 16 + 2 * t + 1] = cv.f[1];
      }
    }
  }

  v8f a00 = {}, a01 = {}, a02 = {}, a03 = {};   // tile 0 (edges e0..e0+15)
  v8f a10 = {}, a11 = {}, a12 = {}, a13 = {};   // tile 1 (edges e0+16..e0+31)
  const _Float16* Bl = Bv + (size_t)lane * 16;

  for (int i = 0; i < 64; ++i) {
    if (i < 63) __builtin_prefetch(Bl + (size_t)((i + 1) * 16) * 512, 0, 0);
    _Float16 xh0 = (_Float16)xs[m * 64 + i];
    _Float16 xh1 = (_Float16)xs[(16 + m) * 64 + i];
#pragma unroll
    for (int kc = 0; kc < 4; ++kc) {
      v16h A0, A1;
#pragma unroll
      for (int j = 0; j < 16; ++j) {
        A0[j] = xh0 * er0[kc * 16 + j];
        A1[j] = xh1 * er1[kc * 16 + j];
      }
      const _Float16* bp = Bl + (size_t)((i * 4 + kc) * 4) * 512;
      v16h b0 = *(const v16h*)(bp);
      v16h b1 = *(const v16h*)(bp + 512);
      v16h b2 = *(const v16h*)(bp + 1024);
      v16h b3 = *(const v16h*)(bp + 1536);
      a00 = wmma_f16(A0, b0, a00);
      a10 = wmma_f16(A1, b0, a10);
      a01 = wmma_f16(A0, b1, a01);
      a11 = wmma_f16(A1, b1, a11);
      a02 = wmma_f16(A0, b2, a02);
      a12 = wmma_f16(A1, b2, a12);
      a03 = wmma_f16(A0, b3, a03);
      a13 = wmma_f16(A1, b3, a13);
    }
  }

  const int n = m;  // C/D col index for this lane
#pragma unroll
  for (int v = 0; v < 8; ++v) {
    int row0 = v + 8 * hi;       // row within tile 0
    int row1 = 16 + row0;        // row within tile 1
    int d0 = dsts[row0];
    int d1 = dsts[row1];
    if (d0 >= 0) {
      atomicAdd(&agg[(size_t)d0 * 64 + 0  + n], a00[v] + nbl[row0 * 64 + 0  + n]);
      atomicAdd(&agg[(size_t)d0 * 64 + 16 + n], a01[v] + nbl[row0 * 64 + 16 + n]);
      atomicAdd(&agg[(size_t)d0 * 64 + 32 + n], a02[v] + nbl[row0 * 64 + 32 + n]);
      atomicAdd(&agg[(size_t)d0 * 64 + 48 + n], a03[v] + nbl[row0 * 64 + 48 + n]);
    }
    if (d1 >= 0) {
      atomicAdd(&agg[(size_t)d1 * 64 + 0  + n], a10[v] + nbl[row1 * 64 + 0  + n]);
      atomicAdd(&agg[(size_t)d1 * 64 + 16 + n], a11[v] + nbl[row1 * 64 + 16 + n]);
      atomicAdd(&agg[(size_t)d1 * 64 + 32 + n], a12[v] + nbl[row1 * 64 + 32 + n]);
      atomicAdd(&agg[(size_t)d1 * 64 + 48 + n], a13[v] + nbl[row1 * 64 + 48 + n]);
    }
  }
}

// ---------------------------------------------------------------------------
// pass1: Y = h @ [Wroot^T | Whh^T]  ->  convlin[N,64] (+bconv), gh[N,192] (+bhh)
// ---------------------------------------------------------------------------
__global__ __launch_bounds__(32) void pass1_k(
    const float* __restrict__ h, const _Float16* __restrict__ Wcatv,
    const float* __restrict__ bconv, const float* __restrict__ bhh,
    float* __restrict__ convlin, float* __restrict__ gh) {
  __shared__ float hs[16 * 64];
  const int lane = threadIdx.x;
  const int n0 = blockIdx.x * 16;
  const float* hp = h + (size_t)n0 * 64;
#pragma unroll
  for (int t = 0; t < 32; ++t) hs[t * 32 + lane] = hp[t * 32 + lane];
  __syncthreads();
  const int m = lane & 15, hi = lane >> 4, n = m;
  v16h A0, A1;
#pragma unroll
  for (int j = 0; j < 16; ++j) {
    int K = (j & 7) + ((j >> 3) << 4) + 8 * hi;
    A0[j] = (_Float16)hs[m * 64 + K];
    A1[j] = (_Float16)hs[m * 64 + 32 + K];
  }
  const _Float16* Wl = Wcatv + (size_t)lane * 16;
#pragma unroll
  for (int nt = 0; nt < 16; ++nt) {
    v8f acc = {};
    acc = wmma_f16(A0, *(const v16h*)(Wl + (size_t)(nt) * 512), acc);
    acc = wmma_f16(A1, *(const v16h*)(Wl + (size_t)(16 + nt) * 512), acc);
#pragma unroll
    for (int v = 0; v < 8; ++v) {
      int o = nt * 16 + n;
      int node = n0 + v + 8 * hi;
      if (o < 64) convlin[(size_t)node * 64 + o] = acc[v] + bconv[o];
      else        gh[(size_t)node * 192 + (o - 64)] = acc[v] + bhh[o - 64];
    }
  }
}

// ---------------------------------------------------------------------------
// pass2: m = silu(agg/deg + convlin); gi = m @ Wih^T (+bih); GRU gates -> h
// ---------------------------------------------------------------------------
__global__ __launch_bounds__(32) void pass2_k(
    const float* __restrict__ agg, const float* __restrict__ deg,
    const float* __restrict__ convlin, const _Float16* __restrict__ Wihv,
    const float* __restrict__ bih, const float* __restrict__ gh,
    float* __restrict__ h) {
  __shared__ float mt[16 * 64];
  const int lane = threadIdx.x;
  const int n0 = blockIdx.x * 16;
#pragma unroll
  for (int t = 0; t < 32; ++t) {
    int idx = t * 32 + lane;
    int row = idx >> 6, c = idx & 63;
    int node = n0 + row;
    float a = agg[(size_t)node * 64 + c] / fmaxf(deg[node], 1.0f) +
              convlin[(size_t)node * 64 + c];
    mt[idx] = dev_silu(a);
  }
  __syncthreads();
  const int m = lane & 15, hi = lane >> 4, n = m;
  v16h A0, A1;
#pragma unroll
  for (int j = 0; j < 16; ++j) {
    int K = (j & 7) + ((j >> 3) << 4) + 8 * hi;
    A0[j] = (_Float16)mt[m * 64 + K];
    A1[j] = (_Float16)mt[m * 64 + 32 + K];
  }
  const _Float16* Wl = Wihv + (size_t)lane * 16;
  v8f g[12];
#pragma unroll
  for (int nt = 0; nt < 12; ++nt) {
    v8f acc = {};
    acc = wmma_f16(A0, *(const v16h*)(Wl + (size_t)(nt) * 512), acc);
    acc = wmma_f16(A1, *(const v16h*)(Wl + (size_t)(12 + nt) * 512), acc);
    g[nt] = acc;
  }
#pragma unroll
  for (int t = 0; t < 4; ++t) {
#pragma unroll
    for (int v = 0; v < 8; ++v) {
      int c = t * 16 + n;
      int node = n0 + v + 8 * hi;
      float ir  = g[t][v]     + bih[c];
      float iz  = g[t + 4][v] + bih[64 + c];
      float in_ = g[t + 8][v] + bih[128 + c];
      const float* gr = gh + (size_t)node * 192;
      float r  = dev_sigmoid(ir + gr[c]);
      float z  = dev_sigmoid(iz + gr[64 + c]);
      float nv = tanhf(in_ + r * gr[128 + c]);
      float ho = h[(size_t)node * 64 + c];
      h[(size_t)node * 64 + c] = (1.0f - z) * nv + z * ho;
    }
  }
}

// ---------------------------------------------------------------------------
// Set2Set
// ---------------------------------------------------------------------------
__global__ void lstm_k(const float* __restrict__ qstar, const float* __restrict__ hL,
                       const float* __restrict__ Wlih, const float* __restrict__ Wlhh,
                       const float* __restrict__ blih, const float* __restrict__ blhh,
                       float* __restrict__ g) {
  int t = blockIdx.x * 256 + threadIdx.x;
  if (t >= NG * 256) return;
  int b = t >> 8, o = t & 255;
  float a = blih[o] + blhh[o];
#pragma unroll 8
  for (int k = 0; k < 128; ++k) a += qstar[b * 128 + k] * Wlih[o * 128 + k];
#pragma unroll 8
  for (int k = 0; k < 64; ++k) a += hL[b * 64 + k] * Wlhh[o * 64 + k];
  g[t] = a;
}

__global__ void lstmgate_k(const float* __restrict__ g, float* __restrict__ cL,
                           float* __restrict__ hL, float* __restrict__ qstar) {
  int t = blockIdx.x * 256 + threadIdx.x;
  if (t >= NG * 64) return;
  int b = t >> 6, c = t & 63;
  const float* gr = g + b * 256;
  float i_ = dev_sigmoid(gr[c]);
  float f_ = dev_sigmoid(gr[64 + c]);
  float gg = tanhf(gr[128 + c]);
  float o_ = dev_sigmoid(gr[192 + c]);
  float cn = f_ * cL[t] + i_ * gg;
  float hn = o_ * tanhf(cn);
  cL[t] = cn;
  hL[t] = hn;
  qstar[b * 128 + c] = hn;  // q-half of qstar for next iter / head
}

__global__ void eatt_k(const float* __restrict__ h, const float* __restrict__ hL,
                       const int* __restrict__ batch, float* __restrict__ eatt,
                       unsigned* __restrict__ emaxU) {
  int n = blockIdx.x * 256 + threadIdx.x;
  if (n >= NN) return;
  int gph = batch[n];
  float a = 0.0f;
#pragma unroll 8
  for (int c = 0; c < 64; ++c) a += h[(size_t)n * 64 + c] * hL[gph * 64 + c];
  eatt[n] = a;
  atomicMax(&emaxU[gph], fenc(a));
}

__global__ void psum_k(const float* __restrict__ eatt, const unsigned* __restrict__ emaxU,
                       const int* __restrict__ batch, float* __restrict__ p,
                       float* __restrict__ asum) {
  int n = blockIdx.x * 256 + threadIdx.x;
  if (n >= NN) return;
  int gph = batch[n];
  float pv = __expf(eatt[n] - fdec(emaxU[gph]));
  p[n] = pv;
  atomicAdd(&asum[gph], pv);
}

__global__ void racc_k(const float* __restrict__ p, const float* __restrict__ asum,
                       const int* __restrict__ batch, const float* __restrict__ h,
                       float* __restrict__ rsum) {
  int t = blockIdx.x * 256 + threadIdx.x;
  if (t >= NN * 64) return;
  int n = t >> 6, c = t & 63;
  int gph = batch[n];
  float coef = p[n] / asum[gph];
  atomicAdd(&rsum[gph * 64 + c], coef * h[t]);
}

__global__ void qstar_r_k(const float* __restrict__ rsum, float* __restrict__ qstar) {
  int t = blockIdx.x * 256 + threadIdx.x;
  if (t >= NG * 64) return;
  int b = t >> 6, c = t & 63;
  qstar[b * 128 + 64 + c] = rsum[t];
}

// ---------------------------------------------------------------------------
// Head
// ---------------------------------------------------------------------------
__global__ void y1_k(const float* __restrict__ qstar, const float* __restrict__ Wo1,
                     const float* __restrict__ bo1, float* __restrict__ y1) {
  int t = blockIdx.x * 256 + threadIdx.x;
  if (t >= NG * 64) return;
  int b = t >> 6, o = t & 63;
  float a = bo1[o];
#pragma unroll 8
  for (int k = 0; k < 128; ++k) a += qstar[b * 128 + k] * Wo1[o * 128 + k];
  y1[t] = a;
}

__global__ void bn2_k(const float* __restrict__ y1, float* __restrict__ bnm,
                      float* __restrict__ bni) {
  int c = threadIdx.x;
  if (c >= 64) return;
  float s = 0.0f, s2 = 0.0f;
  for (int b = 0; b < NG; ++b) {
    float v = y1[b * 64 + c];
    s += v; s2 += v * v;
  }
  float mu = s / (float)NG;
  float var = s2 / (float)NG - mu * mu;
  bnm[c] = mu;
  bni[c] = rsqrtf(var + 1e-5f);
}

__global__ void out_k(const float* __restrict__ y1, const float* __restrict__ bnm,
                      const float* __restrict__ bni, const float* __restrict__ g2,
                      const float* __restrict__ beta2, const float* __restrict__ Wo2,
                      const float* __restrict__ bo2, float* __restrict__ out) {
  int b = blockIdx.x * 256 + threadIdx.x;
  if (b >= NG) return;
  float a = bo2[0];
#pragma unroll 8
  for (int o = 0; o < 64; ++o) {
    float v = (y1[b * 64 + o] - bnm[o]) * bni[o] * g2[o] + beta2[o];
    a += dev_silu(v) * Wo2[o];
  }
  out[b] = a;
}

// ---------------------------------------------------------------------------
// Host orchestration
// ---------------------------------------------------------------------------
extern "C" void kernel_launch(void* const* d_in, const int* in_sizes, int n_in,
                              void* d_out, int out_size, void* d_ws, size_t ws_size,
                              hipStream_t stream) {
  (void)in_sizes; (void)n_in; (void)out_size; (void)ws_size;
  const float* x     = (const float*)d_in[0];
  const float* eattr = (const float*)d_in[1];
  const int*   eidx  = (const int*)d_in[2];
  const int*   batch = (const int*)d_in[3];
  const float* W1    = (const float*)d_in[4];
  const float* b1    = (const float*)d_in[5];
  const float* We1   = (const float*)d_in[6];
  const float* be1   = (const float*)d_in[7];
  const float* g1    = (const float*)d_in[8];
  const float* beta1 = (const float*)d_in[9];
  const float* We2   = (const float*)d_in[10];
  const float* be2   = (const float*)d_in[11];
  const float* Wroot = (const float*)d_in[12];
  const float* bconv = (const float*)d_in[13];
  const float* Wih   = (const float*)d_in[14];
  const float* Whh   = (const float*)d_in[15];
  const float* bih   = (const float*)d_in[16];
  const float* bhh   = (const float*)d_in[17];
  const float* Wlih  = (const float*)d_in[18];
  const float* Wlhh  = (const float*)d_in[19];
  const float* blih  = (const float*)d_in[20];
  const float* blhh  = (const float*)d_in[21];
  const float* Wo1   = (const float*)d_in[22];
  const float* bo1   = (const float*)d_in[23];
  const float* g2    = (const float*)d_in[24];
  const float* beta2 = (const float*)d_in[25];
  const float* Wo2   = (const float*)d_in[26];
  const float* bo2   = (const float*)d_in[27];
  float* out = (float*)d_out;

  // workspace bump allocator (deterministic, local)
  char* w = (char*)d_ws;
  auto alloc = [&](size_t bytes) -> char* {
    char* p = w;
    w += (bytes + 255) & ~((size_t)255);
    return p;
  };
  float*     h       = (float*)alloc((size_t)NN * 64 * 4);
  float*     elin    = (float*)alloc((size_t)EE * 128 * 4);
  _Float16*  eact    = (_Float16*)alloc((size_t)EE * 128 * 2);
  _Float16*  Bv      = (_Float16*)alloc((size_t)BV_ELEMS * 2);
  _Float16*  Wcatv   = (_Float16*)alloc((size_t)WCAT_ELEMS * 2);
  _Float16*  Wihv    = (_Float16*)alloc((size_t)WIH_ELEMS * 2);
  float*     cmean   = (float*)alloc(128 * 4);
  float*     cinv    = (float*)alloc(128 * 4);
  float*     deg     = (float*)alloc((size_t)NN * 4);
  float*     agg     = (float*)alloc((size_t)NN * 64 * 4);
  float*     nb      = (float*)alloc((size_t)NN * 64 * 4);
  float*     convlin = (float*)alloc((size_t)NN * 64 * 4);
  float*     gh      = (float*)alloc((size_t)NN * 192 * 4);
  float*     qstar   = (float*)alloc((size_t)NG * 128 * 4);
  float*     hL      = (float*)alloc((size_t)NG * 64 * 4);
  float*     cL      = (float*)alloc((size_t)NG * 64 * 4);
  float*     glstm   = (float*)alloc((size_t)NG * 256 * 4);
  float*     eatt    = (float*)alloc((size_t)NN * 4);
  float*     pbuf    = (float*)alloc((size_t)NN * 4);
  unsigned*  emaxU   = (unsigned*)alloc((size_t)NG * 4);
  float*     asum    = (float*)alloc((size_t)NG * 4);
  float*     rsum    = (float*)alloc((size_t)NG * 64 * 4);
  float*     y1      = (float*)alloc((size_t)NG * 64 * 4);
  float*     bnm     = (float*)alloc(64 * 4);
  float*     bni     = (float*)alloc(64 * 4);

  auto blocks = [](int n) { return (n + 255) / 256; };

  // ---- preprocessing ----
  prep_k<<<blocks(PREP_TOTAL), 256, 0, stream>>>(We2, Wroot, Whh, Wih, Bv, Wcatv, Wihv);
  embed_k<<<blocks(NN * 64), 256, 0, stream>>>(x, W1, b1, h);
  edgelin_k<<<blocks(EE * 128), 256, 0, stream>>>(eattr, We1, be1, elin);
  colstats_k<<<128, 256, 0, stream>>>(elin, cmean, cinv);
  bnact_k<<<blocks(EE * 128), 256, 0, stream>>>(elin, cmean, cinv, g1, beta1, eact);
  zero_k<<<blocks(NN), 256, 0, stream>>>(deg, NN);
  deg_k<<<blocks(EE), 256, 0, stream>>>(eidx, deg);

  // ---- NNConv + GRU layers ----
  const int etiles = (EE + 31) / 32;
  for (int l = 0; l < LLAY; ++l) {
    nb_k<<<blocks(NN * 64), 256, 0, stream>>>(h, be2, nb);
    zero_k<<<blocks(NN * 64), 256, 0, stream>>>(agg, NN * 64);
    pass1_k<<<NN / 16, 32, 0, stream>>>(h, Wcatv, bconv, bhh, convlin, gh);
    edge_msg_k<<<etiles, 32, 0, stream>>>(h, eact, Bv, nb, eidx, agg);
    pass2_k<<<NN / 16, 32, 0, stream>>>(agg, deg, convlin, Wihv, bih, gh, h);
  }

  // ---- Set2Set ----
  zero_k<<<blocks(NG * 128), 256, 0, stream>>>(qstar, NG * 128);
  zero_k<<<blocks(NG * 64), 256, 0, stream>>>(hL, NG * 64);
  zero_k<<<blocks(NG * 64), 256, 0, stream>>>(cL, NG * 64);
  for (int it = 0; it < MITER; ++it) {
    lstm_k<<<blocks(NG * 256), 256, 0, stream>>>(qstar, hL, Wlih, Wlhh, blih, blhh, glstm);
    lstmgate_k<<<blocks(NG * 64), 256, 0, stream>>>(glstm, cL, hL, qstar);
    zero_k<<<blocks(NG), 256, 0, stream>>>((float*)emaxU, NG);  // 0u == below enc(-inf)
    zero_k<<<blocks(NG), 256, 0, stream>>>(asum, NG);
    zero_k<<<blocks(NG * 64), 256, 0, stream>>>(rsum, NG * 64);
    eatt_k<<<blocks(NN), 256, 0, stream>>>(h, hL, batch, eatt, emaxU);
    psum_k<<<blocks(NN), 256, 0, stream>>>(eatt, emaxU, batch, pbuf, asum);
    racc_k<<<blocks(NN * 64), 256, 0, stream>>>(pbuf, asum, batch, h, rsum);
    qstar_r_k<<<blocks(NG * 64), 256, 0, stream>>>(rsum, qstar);
  }

  // ---- head ----
  y1_k<<<blocks(NG * 64), 256, 0, stream>>>(qstar, Wo1, bo1, y1);
  bn2_k<<<1, 64, 0, stream>>>(y1, bnm, bni);
  out_k<<<1, 256, 0, stream>>>(y1, bnm, bni, g2, beta2, Wo2, bo2, out);
}
